// DyBlock_90572270338247
// MI455X (gfx1250) — matrix operand order repaired and checked by reading, not compiled, not run on hardware
//
#include <hip/hip_runtime.h>
#include <cmath>

// ---------------- problem constants ----------------
#define B_   2
#define L_   512
#define DIM_ 512
#define H_   8
#define H2_  16
#define HD_  32
#define EPS_ 1e-5f

typedef __attribute__((ext_vector_type(16))) _Float16 v16h;
typedef __attribute__((ext_vector_type(8)))  _Float16 v8h;
typedef __attribute__((ext_vector_type(8)))  float    v8f;

#define DEVI __device__ __forceinline__

// ---------------- wave reductions (wave32) ----------------
DEVI float wave_sum(float v) {
#pragma unroll
  for (int o = 16; o > 0; o >>= 1) v += __shfl_xor(v, o, 32);
  return v;
}
DEVI float wave_max(float v) {
#pragma unroll
  for (int o = 16; o > 0; o >>= 1) v = fmaxf(v, __shfl_xor(v, o, 32));
  return v;
}

// ---------------- WMMA fragment loaders (contiguous k = 16*half + i, same for A and B) ----------------
// A fragment: lane holds M = lane&15; K(i) = 16*half + i
//   caller pre-adds row*lda + 16*half; two aligned 16-byte chunks.
DEVI v16h frag_a_load(const _Float16* pa) {
  v8h lo = *(const v8h*)pa;
  v8h hi = *(const v8h*)(pa + 8);
  v16h f;
#pragma unroll
  for (int i = 0; i < 8; ++i) { f[i] = lo[i]; f[8 + i] = hi[i]; }
  return f;
}
// fused differential attention: attn0 - lam*attn1 in packed f16
DEVI v16h frag_a_diffload(const _Float16* pa, const _Float16* pb, _Float16 lamh) {
  v8h a0 = *(const v8h*)pa, a1 = *(const v8h*)(pa + 8);
  v8h b0 = *(const v8h*)pb, b1 = *(const v8h*)(pb + 8);
  v16h f;
#pragma unroll
  for (int i = 0; i < 8; ++i) {
    f[i]     = a0[i] - lamh * b0[i];
    f[8 + i] = a1[i] - lamh * b1[i];
  }
  return f;
}

enum { AM_RM = 0, AM_DIFF = 1 };
enum { EP_PLAIN = 0, EP_SCORES = 1, EP_ADD = 2, EP_SWIGLU = 3, EP_ZOUT = 4 };

template <int AM>
DEVI void load_afrags(v16h a[2], const _Float16* r0, const _Float16* r1,
                      long plane, _Float16 lamh, int kk) {
  if (AM == AM_RM) {
    a[0] = frag_a_load(r0 + kk);
    a[1] = frag_a_load(r1 + kk);
  } else {
    a[0] = frag_a_diffload(r0 + kk, r0 + kk + plane, lamh);
    a[1] = frag_a_diffload(r1 + kk, r1 + kk + plane, lamh);
  }
}
DEVI void load_bfrags(v16h b[4], const _Float16* const brow[4], int kk) {
#pragma unroll
  for (int t = 0; t < 4; ++t) b[t] = *(const v16h*)(brow[t] + kk);
}
DEVI void mma8(v8f acc[2][4], const v16h a[2], const v16h b[4]) {
#pragma unroll
  for (int u = 0; u < 2; ++u) {
#pragma unroll
    for (int t = 0; t < 4; ++t) {
      acc[u][t] = __builtin_amdgcn_wmma_f32_16x16x32_f16(
          false, a[u], false, b[t], (short)0, acc[u][t], false, false);
    }
  }
}

// ---------------- generic batched WMMA GEMM (f16 operands, f32 accum) ------
struct GemmP {
  const void* A; const void* B; void* C;
  const float* X1; const float* X2; const float* lamp;
  long lda, ldb, crow, ccol;                 // element (half / float) strides
  long sA1, sA2, sB1, sB2, sC1, sC2, sX11, sX12, sX21, sX22;
  long x1row, x1col;
  long plane;                                // AM_DIFF: offset (halves) to 2nd plane
  int  M, N, K, zdiv;
  float scale;
};

// block = 256 threads = 8 waves; wave computes 32(M) x 64(N); block = 256 x 64.
// K-loop manually unrolled x2 with two named fragment sets (zero-copy ping-pong):
// set1 loads overlap set0 WMMAs and vice versa.
template <int AM, int EP, int OUT16>
__global__ __launch_bounds__(256) void k_gemm(GemmP p) {
  const int z = blockIdx.z, z1 = z / p.zdiv, z2 = z % p.zdiv;
  const _Float16* A  = (const _Float16*)p.A + (long)z1 * p.sA1 + (long)z2 * p.sA2;
  const _Float16* Bm = (const _Float16*)p.B + (long)z1 * p.sB1 + (long)z2 * p.sB2;
  const long coff = (long)z1 * p.sC1 + (long)z2 * p.sC2;
  const float* X1 = p.X1 ? p.X1 + (long)z1 * p.sX11 + (long)z2 * p.sX12 : nullptr;
  const float* X2 = p.X2 ? p.X2 + (long)z1 * p.sX21 + (long)z2 * p.sX22 : nullptr;

  const int lane = threadIdx.x & 31;
  const int wv   = threadIdx.x >> 5;
  const int nl   = lane & 15;
  const int half = lane >> 4;
  const int m0 = blockIdx.y * 256 + wv * 32;
  const int n0 = blockIdx.x * 64;
  if (m0 >= p.M) return;

  const _Float16 lamh = (AM == AM_DIFF) ? (_Float16)p.lamp[0] : (_Float16)0.0f;
  const _Float16* arow0 = A + (long)(m0 + nl) * p.lda + 16 * half;
  const _Float16* arow1 = arow0 + 16 * p.lda;
  const _Float16* brow[4];
#pragma unroll
  for (int t = 0; t < 4; ++t)
    brow[t] = Bm + (long)(n0 + t * 16 + nl) * p.ldb + 16 * half;

  const int S = p.K >> 5;  // number of K=32 steps (>= 1)

  v16h a0[2], b0[4], a1[2], b1[4];
  load_afrags<AM>(a0, arow0, arow1, p.plane, lamh, 0);
  load_bfrags(b0, brow, 0);

  v8f acc[2][4] = {};

  int s = 0;
  for (; s + 2 <= S; s += 2) {
    const int k1 = (s + 1) << 5;
    __builtin_prefetch(arow0 + k1 + 32, 0, 1);  // global_prefetch_b8
    load_afrags<AM>(a1, arow0, arow1, p.plane, lamh, k1);
    load_bfrags(b1, brow, k1);
    mma8(acc, a0, b0);                           // overlaps set1 loads
    if (s + 2 < S) {
      const int k2 = (s + 2) << 5;
      load_afrags<AM>(a0, arow0, arow1, p.plane, lamh, k2);
      load_bfrags(b0, brow, k2);
    }
    mma8(acc, a1, b1);                           // overlaps set0 reloads
  }
  if (s < S) mma8(acc, a0, b0);                  // odd tail (e.g. K == 32)

  // epilogue: C/D layout — lane holds N = nl; VGPR r holds M = 16*u + 8*half + r
#pragma unroll
  for (int u = 0; u < 2; ++u) {
#pragma unroll
    for (int t = 0; t < 4; ++t) {
#pragma unroll
      for (int r = 0; r < 8; ++r) {
        const long m = m0 + u * 16 + 8 * half + r;
        const long n = n0 + t * 16 + nl;
        float vacc = acc[u][t][r];
        float res;
        if (EP == EP_PLAIN) {
          res = vacc;
        } else if (EP == EP_SCORES) {
          res = vacc * p.scale + X1[m * p.x1row + n * p.x1col];
        } else if (EP == EP_ADD) {
          res = vacc + X1[m * p.x1row + n * p.x1col];
        } else if (EP == EP_SWIGLU) {
          float xpv = X2[m * p.x1row + n * p.x1col];
          float sil = xpv / (1.0f + __expf(-xpv));
          res = X1[m * p.x1row + n * p.x1col] + vacc * sil;
        } else {  // EP_ZOUT: tn + z + bout[o]
          res = vacc + X1[m * p.x1row + n * p.x1col] + X2[0];
        }
        const long ci = coff + m * p.crow + n * p.ccol;
        if (OUT16) ((_Float16*)p.C)[ci] = (_Float16)res;
        else       ((float*)p.C)[ci]    = res;
      }
    }
  }
}

// ---------------- LDS-tiled transpose with f32 -> f16 convert ----------------
__global__ __launch_bounds__(256) void k_transpose_cvt(
    const float* in, long inRS, long inBS1, long inBS2,
    _Float16* out, long outRS, long outBS1, long outBS2,
    int R, int C, int zdiv) {
  __shared__ float tile[32][33];
  const int z = blockIdx.z, z1 = z / zdiv, z2 = z % zdiv;
  const float* I = in + (long)z1 * inBS1 + (long)z2 * inBS2;
  _Float16*    O = out + (long)z1 * outBS1 + (long)z2 * outBS2;
  const int r0 = blockIdx.y * 32, c0 = blockIdx.x * 32;
  const int tx = threadIdx.x, ty = threadIdx.y;
#pragma unroll
  for (int i = 0; i < 32; i += 8) {
    int r = r0 + ty + i, c = c0 + tx;
    tile[ty + i][tx] = (r < R && c < C) ? I[(long)r * inRS + c] : 0.f;
  }
  __syncthreads();
#pragma unroll
  for (int i = 0; i < 32; i += 8) {
    int c = c0 + ty + i, r = r0 + tx;
    if (c < C && r < R) O[(long)c * outRS + r] = (_Float16)tile[tx][ty + i];
  }
}

// ---------------- elementwise / norm kernels ----------------
__global__ void k_cvt(const float* in, _Float16* out, int n) {
  int i = blockIdx.x * blockDim.x + threadIdx.x;
  if (i < n) out[i] = (_Float16)in[i];
}

__global__ void k_rms_rows_h(const float* in, _Float16* out, int nrows, int len) {
  int row = (int)((blockIdx.x * blockDim.x + threadIdx.x) >> 5);
  if (row >= nrows) return;
  int lane = threadIdx.x & 31;
  const float* r = in + (size_t)row * len;
  float ss = 0.f;
  for (int i = lane; i < len; i += 32) { float v = r[i]; ss += v * v; }
  ss = wave_sum(ss);
  float inv = rsqrtf(ss / (float)len + EPS_);
  _Float16* w = out + (size_t)row * len;
  for (int i = lane; i < len; i += 32) w[i] = (_Float16)(r[i] * inv);
}

__global__ void k_rms_t(const float* t, float* tn, long nrows) {
  long row = blockIdx.x * (long)blockDim.x + threadIdx.x;
  if (row >= nrows) return;
  const float4* p = (const float4*)(t + row * 16);
  float4 q[4];
  float ss = 0.f;
#pragma unroll
  for (int i = 0; i < 4; ++i) {
    q[i] = p[i];
    ss += q[i].x * q[i].x + q[i].y * q[i].y + q[i].z * q[i].z + q[i].w * q[i].w;
  }
  float inv = rsqrtf(ss / 16.f + EPS_);
  float4* w = (float4*)(tn + row * 16);
#pragma unroll
  for (int i = 0; i < 4; ++i) {
    float4 v = q[i];
    v.x *= inv; v.y *= inv; v.z *= inv; v.w *= inv;
    w[i] = v;
  }
}

// in-place f16 softmax over rows of 512
__global__ void k_softmax_h(_Float16* attn, int nrows) {
  int row = (int)((blockIdx.x * blockDim.x + threadIdx.x) >> 5);
  if (row >= nrows) return;
  int lane = threadIdx.x & 31;
  _Float16* r = attn + (size_t)row * 512;
  float v[16];
  float mx = -1e30f;
#pragma unroll
  for (int i = 0; i < 16; ++i) { v[i] = (float)r[lane + 32 * i]; mx = fmaxf(mx, v[i]); }
  mx = wave_max(mx);
  float s = 0.f;
#pragma unroll
  for (int i = 0; i < 16; ++i) { v[i] = __expf(v[i] - mx); s += v[i]; }
  s = wave_sum(s);
  float inv = 1.f / s;
#pragma unroll
  for (int i = 0; i < 16; ++i) r[lane + 32 * i] = (_Float16)(v[i] * inv);
}

__global__ void k_rms_o(const float* o, _Float16* o16, int nrows, float postscale) {
  int row = (int)((blockIdx.x * blockDim.x + threadIdx.x) >> 5);
  if (row >= nrows) return;
  int lane = threadIdx.x & 31;
  const float* r = o + (size_t)row * 64;
  float a = r[lane], b = r[lane + 32];
  float ss = wave_sum(a * a + b * b);
  float inv = rsqrtf(ss / 64.f + EPS_) * postscale;
  _Float16* w = o16 + (size_t)row * 64;
  w[lane] = (_Float16)(a * inv);
  w[lane + 32] = (_Float16)(b * inv);
}

__global__ void k_lnorm_h(const float* in, _Float16* out, const float* g, const float* b,
                          int nrows, int len) {
  int row = (int)((blockIdx.x * blockDim.x + threadIdx.x) >> 5);
  if (row >= nrows) return;
  int lane = threadIdx.x & 31;
  const float* r = in + (size_t)row * len;
  float s = 0.f, ss = 0.f;
  for (int i = lane; i < len; i += 32) { float xv = r[i]; s += xv; ss += xv * xv; }
  s = wave_sum(s); ss = wave_sum(ss);
  float mu = s / (float)len;
  float var = ss / (float)len - mu * mu;
  float inv = rsqrtf(var + EPS_);
  _Float16* w = out + (size_t)row * len;
  for (int i = lane; i < len; i += 32) w[i] = (_Float16)((r[i] - mu) * inv * g[i] + b[i]);
}

__global__ void k_lambda(const float* lq1, const float* lk1,
                         const float* lq2, const float* lk2,
                         float* lam, float lambda_init) {
  if (threadIdx.x == 0) {
    float d1 = 0.f, d2 = 0.f;
    for (int i = 0; i < HD_; ++i) { d1 += lq1[i] * lk1[i]; d2 += lq2[i] * lk2[i]; }
    *lam = __expf(d1) - __expf(d2) + lambda_init;
  }
}

// ---------------- host orchestration ----------------
extern "C" void kernel_launch(void* const* d_in, const int* in_sizes, int n_in,
                              void* d_out, int out_size, void* d_ws, size_t ws_size,
                              hipStream_t stream) {
  (void)in_sizes; (void)n_in; (void)out_size; (void)ws_size;

  const float* x    = (const float*)d_in[0];
  const float* t    = (const float*)d_in[1];
  const float* Wq   = (const float*)d_in[2];
  const float* Wk   = (const float*)d_in[3];
  const float* Wv   = (const float*)d_in[4];
  const float* Wo   = (const float*)d_in[5];
  const float* lq1  = (const float*)d_in[6];
  const float* lk1  = (const float*)d_in[7];
  const float* lq2  = (const float*)d_in[8];
  const float* lk2  = (const float*)d_in[9];
  const float* W1   = (const float*)d_in[10];
  const float* W2   = (const float*)d_in[11];
  const float* ln_g = (const float*)d_in[12];
  const float* ln_b = (const float*)d_in[13];
  const float* Wa   = (const float*)d_in[14];
  const float* Wb   = (const float*)d_in[15];
  const float* Wout = (const float*)d_in[16];
  const float* bout = (const float*)d_in[17];

  const size_t XN  = (size_t)B_ * L_ * DIM_;        // 524288
  const size_t TSZ = (size_t)B_ * L_ * L_ * H2_;    // 8388608
  const size_t WSZ = (size_t)DIM_ * DIM_;           // 262144

  float* xout = (float*)d_out;
  float* tout = (float*)d_out + XN;

  // ---- workspace carve (f16 buffers occupy half a float each) ----
  float* ws   = (float*)d_ws;
  float* tn   = ws; ws += TSZ;
  float* ybuf = ws; ws += XN;
  float* xpb  = ws; ws += XN;
  float* obuf = ws; ws += XN;
  float* vbuf = ws; ws += XN;
  float* lamp = ws; ws += 64;
  _Float16* xn16   = (_Float16*)ws; ws += XN / 2;
  _Float16* q16    = (_Float16*)ws; ws += XN / 2;
  _Float16* k16    = (_Float16*)ws; ws += XN / 2;
  _Float16* vT16   = (_Float16*)ws; ws += XN / 2;   // [b,h,e,j]
  _Float16* o16    = (_Float16*)ws; ws += XN / 2;
  _Float16* yn16   = (_Float16*)ws; ws += XN / 2;
  _Float16* xln16  = (_Float16*)ws; ws += XN / 2;
  _Float16* aa16   = (_Float16*)ws; ws += XN / 2;
  _Float16* bb16   = (_Float16*)ws; ws += XN / 2;
  _Float16* attn16 = (_Float16*)ws; ws += TSZ / 2;  // scores -> softmax in place
  _Float16* tmp16  = (_Float16*)ws; ws += TSZ / 2;  // [b,i,o,d]
  _Float16* Wq16   = (_Float16*)ws; ws += WSZ / 2;
  _Float16* Wk16   = (_Float16*)ws; ws += WSZ / 2;
  _Float16* Wv16   = (_Float16*)ws; ws += WSZ / 2;
  _Float16* Wo16   = (_Float16*)ws; ws += WSZ / 2;
  _Float16* W116   = (_Float16*)ws; ws += WSZ / 2;
  _Float16* W216   = (_Float16*)ws; ws += WSZ / 2;
  _Float16* Wa16   = (_Float16*)ws; ws += WSZ / 2;
  _Float16* Wb16   = (_Float16*)ws; ws += WSZ / 2;
  _Float16* WoutT16 = (_Float16*)ws; ws += ((size_t)H2_ * WSZ) / 2;  // [o,d,c]

  const float LAMBDA_INIT = 0.8f - 0.6f * expf(-0.3f * 11.0f);
  const dim3 thr(256);

  // ---- one-shot conversions + norms + lambda ----
  k_cvt<<<dim3(1024), thr, 0, stream>>>(Wq, Wq16, (int)WSZ);
  k_cvt<<<dim3(1024), thr, 0, stream>>>(Wk, Wk16, (int)WSZ);
  k_cvt<<<dim3(1024), thr, 0, stream>>>(Wv, Wv16, (int)WSZ);
  k_cvt<<<dim3(1024), thr, 0, stream>>>(Wo, Wo16, (int)WSZ);
  k_cvt<<<dim3(1024), thr, 0, stream>>>(W1, W116, (int)WSZ);
  k_cvt<<<dim3(1024), thr, 0, stream>>>(W2, W216, (int)WSZ);
  k_cvt<<<dim3(1024), thr, 0, stream>>>(Wa, Wa16, (int)WSZ);
  k_cvt<<<dim3(1024), thr, 0, stream>>>(Wb, Wb16, (int)WSZ);
  // WoutT[o,d,c] = Wout[o,c,d]
  k_transpose_cvt<<<dim3(16, 16, H2_), dim3(32, 8), 0, stream>>>(
      Wout, DIM_, 0, (long)WSZ, WoutT16, DIM_, 0, (long)WSZ, DIM_, DIM_, H2_);

  k_rms_rows_h<<<dim3(128), thr, 0, stream>>>(x, xn16, B_ * L_, DIM_);
  k_rms_t<<<dim3(2048), thr, 0, stream>>>(t, tn, (long)B_ * L_ * L_);
  k_lambda<<<dim3(1), dim3(32), 0, stream>>>(lq1, lk1, lq2, lk2, lamp, LAMBDA_INIT);

  // ---- q/k/v projections: xn @ W.T ----
  {
    GemmP p{}; p.zdiv = 1; p.scale = 1.f;
    p.A = xn16; p.lda = DIM_; p.ldb = DIM_;
    p.crow = DIM_; p.ccol = 1;
    p.M = B_ * L_; p.N = DIM_; p.K = DIM_;
    p.B = Wq16; p.C = q16;
    k_gemm<AM_RM, EP_PLAIN, 1><<<dim3(8, 4, 1), thr, 0, stream>>>(p);
    p.B = Wk16; p.C = k16;
    k_gemm<AM_RM, EP_PLAIN, 1><<<dim3(8, 4, 1), thr, 0, stream>>>(p);
    p.B = Wv16; p.C = vbuf;
    k_gemm<AM_RM, EP_PLAIN, 0><<<dim3(8, 4, 1), thr, 0, stream>>>(p);
  }
  // vT16[b,h,e,j] = v[b,j,h,e]
  k_transpose_cvt<<<dim3(2, 16, B_ * H_), dim3(32, 8), 0, stream>>>(
      vbuf, DIM_, (long)L_ * DIM_, 2 * HD_,
      vT16, L_, (long)H_ * 2 * HD_ * L_, (long)2 * HD_ * L_, L_, 2 * HD_, H_);

  // ---- scores = q_h k_h^T / sqrt(HD) + tn  -> f16, batch over (b,h2) ----
  {
    GemmP p{}; p.zdiv = H2_; p.scale = 0.17677669529663687f;  // 1/sqrt(32)
    p.A = q16; p.lda = DIM_; p.sA1 = (long)L_ * DIM_; p.sA2 = HD_;
    p.B = k16; p.ldb = DIM_; p.sB1 = (long)L_ * DIM_; p.sB2 = HD_;
    p.C = attn16; p.crow = L_; p.ccol = 1;
    p.sC1 = (long)H2_ * L_ * L_; p.sC2 = (long)L_ * L_;
    p.X1 = tn; p.x1row = (long)L_ * H2_; p.x1col = H2_;
    p.sX11 = (long)L_ * L_ * H2_; p.sX12 = 1;
    p.M = L_; p.N = L_; p.K = HD_;
    k_gemm<AM_RM, EP_SCORES, 1><<<dim3(8, 2, B_ * H2_), thr, 0, stream>>>(p);
  }

  // ---- softmax rows (f16 in place) ----
  k_softmax_h<<<dim3(2048), thr, 0, stream>>>(attn16, B_ * H2_ * L_);

  // ---- o = (attn0 - lam*attn1) @ v_h  (diff fused in f16; batch over b,h) ----
  {
    GemmP p{}; p.zdiv = H_; p.scale = 1.f;
    p.A = attn16; p.lda = L_; p.plane = (long)L_ * L_;
    p.sA1 = (long)H2_ * L_ * L_; p.sA2 = 2L * L_ * L_;
    p.lamp = lamp;
    p.B = vT16; p.ldb = L_;
    p.sB1 = (long)H_ * 2 * HD_ * L_; p.sB2 = (long)2 * HD_ * L_;
    p.C = obuf; p.crow = DIM_; p.ccol = 1;
    p.sC1 = (long)L_ * DIM_; p.sC2 = 2 * HD_;
    p.M = L_; p.N = 2 * HD_; p.K = L_;
    k_gemm<AM_DIFF, EP_PLAIN, 0><<<dim3(1, 2, B_ * H_), thr, 0, stream>>>(p);
  }

  // ---- per-head rmsnorm(o) * (1 - lambda_init) -> f16 ----
  k_rms_o<<<dim3(1024), thr, 0, stream>>>(obuf, o16, B_ * L_ * H_, 1.0f - LAMBDA_INIT);

  // ---- y = x + o @ Wo.T ----
  {
    GemmP p{}; p.zdiv = 1; p.scale = 1.f;
    p.A = o16; p.lda = DIM_; p.B = Wo16; p.ldb = DIM_;
    p.C = ybuf; p.crow = DIM_; p.ccol = 1;
    p.X1 = x; p.x1row = DIM_; p.x1col = 1;
    p.M = B_ * L_; p.N = DIM_; p.K = DIM_;
    k_gemm<AM_RM, EP_ADD, 0><<<dim3(8, 4, 1), thr, 0, stream>>>(p);
  }

  // ---- SwiGLU ----
  k_rms_rows_h<<<dim3(128), thr, 0, stream>>>(ybuf, yn16, B_ * L_, DIM_);
  {
    GemmP p{}; p.zdiv = 1; p.scale = 1.f;
    p.A = yn16; p.lda = DIM_; p.B = W216; p.ldb = DIM_;
    p.C = xpb; p.crow = DIM_; p.ccol = 1;
    p.M = B_ * L_; p.N = DIM_; p.K = DIM_;
    k_gemm<AM_RM, EP_PLAIN, 0><<<dim3(8, 4, 1), thr, 0, stream>>>(p);
    p.B = W116; p.C = xout;
    p.X1 = ybuf; p.X2 = xpb; p.x1row = DIM_; p.x1col = 1;
    k_gemm<AM_RM, EP_SWIGLU, 0><<<dim3(8, 4, 1), thr, 0, stream>>>(p);
  }

  // ---- layernorm(x_out) -> f16; aa/bb projections ----
  k_lnorm_h<<<dim3(128), thr, 0, stream>>>(xout, xln16, ln_g, ln_b, B_ * L_, DIM_);
  {
    GemmP p{}; p.zdiv = 1; p.scale = 1.f;
    p.A = xln16; p.lda = DIM_; p.ldb = DIM_;
    p.crow = DIM_; p.ccol = 1;
    p.M = B_ * L_; p.N = DIM_; p.K = DIM_;
    p.B = Wa16; p.C = aa16;
    k_gemm<AM_RM, EP_PLAIN, 1><<<dim3(8, 4, 1), thr, 0, stream>>>(p);
    p.B = Wb16; p.C = bb16;
    k_gemm<AM_RM, EP_PLAIN, 1><<<dim3(8, 4, 1), thr, 0, stream>>>(p);
  }

  // ---- tmp[b,i,o,:] = aa[b,i,:] @ WoutT[o].T  (batch over o) ----
  {
    GemmP p{}; p.zdiv = H2_; p.scale = 1.f;
    p.A = aa16; p.lda = DIM_;
    p.B = WoutT16; p.ldb = DIM_; p.sB2 = (long)DIM_ * DIM_;
    p.C = tmp16; p.crow = (long)H2_ * DIM_; p.ccol = 1; p.sC2 = DIM_;
    p.M = B_ * L_; p.N = DIM_; p.K = DIM_;
    k_gemm<AM_RM, EP_PLAIN, 1><<<dim3(8, 4, H2_), thr, 0, stream>>>(p);
  }

  // ---- t_out[b,i,j,o] = tn + tmp[b,i,o,:].bb[b,j,:] + bout[o]  (batch b,o) ----
  {
    GemmP p{}; p.zdiv = H2_; p.scale = 1.f;
    p.A = tmp16; p.lda = (long)H2_ * DIM_;
    p.sA1 = (long)L_ * H2_ * DIM_; p.sA2 = DIM_;
    p.B = bb16; p.ldb = DIM_; p.sB1 = (long)L_ * DIM_; p.sB2 = 0;
    p.C = tout; p.crow = (long)L_ * H2_; p.ccol = H2_;
    p.sC1 = (long)L_ * L_ * H2_; p.sC2 = 1;
    p.X1 = tn; p.x1row = (long)L_ * H2_; p.x1col = H2_;
    p.sX11 = (long)L_ * L_ * H2_; p.sX12 = 1;
    p.X2 = bout; p.sX21 = 0; p.sX22 = 1;
    p.M = L_; p.N = L_; p.K = DIM_;
    k_gemm<AM_RM, EP_ZOUT, 0><<<dim3(8, 2, B_ * H2_), thr, 0, stream>>>(p);
  }
}
